// BEVFeatureGenerator_85289460564572
// MI455X (gfx1250) — matrix operand order repaired and checked by reading, not compiled, not run on hardware
//
#include <hip/hip_runtime.h>

// ---------------------------------------------------------------------------
// BEV feature generator for MI455X (gfx1250, wave32, WMMA).
//
// Per 16-point tile (one wave):
//   fused[16x128] = concat(pos_gather, obj_gather)   (f32 -> f16, LDS)
//   hid  [16x128] = relu(fused @ W1 + b1)            (v_wmma_f32_16x16x32_f16)
//   out  [16x64]  = hid @ W2 + b2                    (v_wmma_f32_16x16x32_f16)
//   scatter out rows to bev[y,x,:]
//
// Weights are converted f32->f16 ONCE per block into LDS (transposed,
// XOR-swizzled 16B chunks so fragment loads hit all 64 banks), then each
// wave loops over 8 tiles (128 points) to amortize the staging cost.
// Memory-bound workload: ~0.8 GB of HBM traffic vs 51.5 GFLOP of f16 WMMA.
// ---------------------------------------------------------------------------

typedef __attribute__((ext_vector_type(16))) _Float16 v16h;
typedef __attribute__((ext_vector_type(8)))  float    v8f;

#define HGT      1024
#define WID      1024
#define OUT_DIM  64
#define NPOINTS  1048576
#define WAVES    4
#define TPB      (WAVES * 32)
#define TILES_PER_WAVE 8
#define PTS_PER_WAVE   (TILES_PER_WAVE * 16)     // 128
#define PTS_PER_BLOCK  (WAVES * PTS_PER_WAVE)    // 512

// LDS layout (dynamic, 65536 B total):
//   [0,      32768)  W1^T : 128 rows (hidden n) x 128 f16 (k), swizzled
//   [32768,  49152)  W2^T :  64 rows (out    n) x 128 f16 (k), swizzled
//   [49152,  65536)  per-wave 16 x 128 f16 tile (fused, then reused for hid)
#define LDS_W1T   0
#define LDS_W2T   32768
#define LDS_TILE  49152
#define TILE_BYTES 4096   // 16 rows * 256 B

static __device__ inline int swzoff(int row, int chunk) {
  // row stride 256 B (128 halves); 16B chunk index XOR-swizzled by row.
  return row * 256 + (((chunk ^ row) & 15) << 4);
}

static __device__ inline unsigned short f2h_bits(float f) {
  union { _Float16 h; unsigned short u; } x;
  x.h = (_Float16)f;
  return x.u;
}

// A fragment, 16x32 f16 (ISA 7.12.2): lane<16 -> row M=lane, K = {b..b+7, b+16..b+23};
// lane>=16 -> row M=lane-16, K = {b+8..b+15, b+24..b+31}; b = 32*kb.
static __device__ inline v16h load_afrag(const char* base, int kb, int lane) {
  int row = lane & 15;
  int o   = lane >> 4;            // 0 | 1
  int c0  = 4 * kb + o;
  union { v16h v; uint4 q[2]; } u;
  u.q[0] = *(const uint4*)(base + swzoff(row, c0));
  u.q[1] = *(const uint4*)(base + swzoff(row, c0 + 2));
  return u.v;
}

// B fragment, 32x16 f16: lane holds column N = rowbase + (lane&15),
// K = {b+8o..b+8o+7, b+16+8o..b+16+8o+7}; b = 32*kb, o = lane>>4 (matches A).
static __device__ inline v16h load_bfrag(const char* base, int rowbase, int kb, int lane) {
  int row = rowbase + (lane & 15);
  int o   = lane >> 4;
  int c0  = 4 * kb + o;
  union { v16h v; uint4 q[2]; } u;
  u.q[0] = *(const uint4*)(base + swzoff(row, c0));
  u.q[1] = *(const uint4*)(base + swzoff(row, c0 + 2));
  return u.v;
}

__global__ void __launch_bounds__(256) bev_zero_kernel(float4* out, int n4) {
  int i = blockIdx.x * blockDim.x + threadIdx.x;
  if (i < n4) out[i] = float4{0.f, 0.f, 0.f, 0.f};
}

__global__ void __launch_bounds__(TPB) bev_wmma_kernel(
    const float* __restrict__ pos,    // [1024*1024*64]
    const float* __restrict__ obj,    // [4096*64]
    const float* __restrict__ w1,     // [128*128] (w1[k*128+n])
    const float* __restrict__ b1,     // [128]
    const float* __restrict__ w2,     // [128*64]  (w2[k*64+n])
    const float* __restrict__ b2,     // [64]
    const int*   __restrict__ grids,  // [N,2] (y,x)
    const int*   __restrict__ ids,    // [N]
    float*       __restrict__ bev)    // [1024*1024*64]
{
  extern __shared__ char smem[];
  const int tid    = threadIdx.x;
  const int lane   = tid & 31;
  const int waveId = tid >> 5;
  char* const w1t  = smem + LDS_W1T;
  char* const w2t  = smem + LDS_W2T;
  char* const tile = smem + LDS_TILE + waveId * TILE_BYTES;

  const int waveBase = blockIdx.x * PTS_PER_BLOCK + waveId * PTS_PER_WAVE;

  // Warm the cache path for this wave's first tile metadata while weights stage.
  __builtin_prefetch(grids + 2 * waveBase + lane, 0, 3);

  // ---- stage W1^T into LDS as f16 (coalesced global reads: n = tid fast) ----
  #pragma unroll 4
  for (int k = 0; k < 128; ++k) {
    float v = w1[k * 128 + tid];
    *(unsigned short*)(w1t + swzoff(tid, k >> 3) + ((k & 7) << 1)) = f2h_bits(v);
  }
  // ---- stage W2^T ----
  #pragma unroll 4
  for (int it = 0; it < 64; ++it) {
    int flat = it * 128 + tid;
    int n = flat & 63, k = flat >> 6;
    float v = w2[k * 64 + n];
    *(unsigned short*)(w2t + swzoff(n, k >> 3) + ((k & 7) << 1)) = f2h_bits(v);
  }
  __syncthreads();

  // ---- tile-loop-invariant bias values (per lane) ----
  const int mbase = (lane >> 4) << 3;   // f32 C-fragment row base: 0 | 8
  float bias1[8], bias2[4];
  #pragma unroll
  for (int nt = 0; nt < 8; ++nt) bias1[nt] = b1[nt * 16 + (lane & 15)];
  #pragma unroll
  for (int nt = 0; nt < 4; ++nt) bias2[nt] = b2[nt * 16 + (lane & 15)];

  for (int t = 0; t < TILES_PER_WAVE; ++t) {
    const int tile0 = waveBase + t * 16;

    // Prefetch next tile's metadata while this tile computes.
    if (t + 1 < TILES_PER_WAVE)
      __builtin_prefetch(grids + 2 * (tile0 + 16) + lane, 0, 3);

    // ---- per-point metadata (lanes 0..15 own one point each) ----
    int cellOff = 0, objOff = 0;
    if (lane < 16) {
      int p = tile0 + lane;
      int y = grids[2 * p];
      int x = grids[2 * p + 1];
      cellOff = (y * WID + x) * OUT_DIM;   // f32 offset of bev cell / pos row
      objOff  = ids[p] * 64;
    }

    // ---- gather fused features -> LDS f16 (rows = points, 128 cols) ----
    #pragma unroll
    for (int m = 0; m < 16; ++m) {
      int cOff = __shfl(cellOff, m);
      int oOff = __shfl(objOff, m);
      float4 f;
      if (lane < 16) f = ((const float4*)(pos + cOff))[lane];        // pos feats 0..63
      else           f = ((const float4*)(obj + oOff))[lane - 16];   // obj feats 64..127
      int col = (lane < 16) ? (lane << 2) : (64 + ((lane - 16) << 2));
      union { uint2 d; unsigned short s[4]; } pk;
      pk.s[0] = f2h_bits(f.x); pk.s[1] = f2h_bits(f.y);
      pk.s[2] = f2h_bits(f.z); pk.s[3] = f2h_bits(f.w);
      *(uint2*)(tile + swzoff(m, col >> 3) + ((col & 7) << 1)) = pk.d;
    }

    // ---- GEMM1: hid = relu(fused @ W1 + b1) ----
    v16h A[4];
    #pragma unroll
    for (int kb = 0; kb < 4; ++kb) A[kb] = load_afrag(tile, kb, lane);

    #pragma unroll
    for (int nt = 0; nt < 8; ++nt) {
      v8f c = {0.f, 0.f, 0.f, 0.f, 0.f, 0.f, 0.f, 0.f};
      #pragma unroll
      for (int kb = 0; kb < 4; ++kb) {
        v16h b = load_bfrag(w1t, nt * 16, kb, lane);
        c = __builtin_amdgcn_wmma_f32_16x16x32_f16(false, A[kb], false, b,
                                                   (short)0, c, false, false);
      }
      int hcol = nt * 16 + (lane & 15);
      #pragma unroll
      for (int r = 0; r < 8; ++r) {
        float v = c[r] + bias1[nt];
        v = v > 0.f ? v : 0.f;
        int row = mbase + r;
        *(unsigned short*)(tile + swzoff(row, hcol >> 3) + ((hcol & 7) << 1)) = f2h_bits(v);
      }
    }

    // ---- GEMM2: out = hid @ W2 + b2, scatter to bev ----
    v16h A2[4];
    #pragma unroll
    for (int kb = 0; kb < 4; ++kb) A2[kb] = load_afrag(tile, kb, lane);

    int cellRow[8];
    #pragma unroll
    for (int r = 0; r < 8; ++r) cellRow[r] = __shfl(cellOff, mbase + r);

    #pragma unroll
    for (int nt = 0; nt < 4; ++nt) {
      v8f c = {0.f, 0.f, 0.f, 0.f, 0.f, 0.f, 0.f, 0.f};
      #pragma unroll
      for (int kb = 0; kb < 4; ++kb) {
        v16h b = load_bfrag(w2t, nt * 16, kb, lane);
        c = __builtin_amdgcn_wmma_f32_16x16x32_f16(false, A2[kb], false, b,
                                                   (short)0, c, false, false);
      }
      int col = nt * 16 + (lane & 15);
      #pragma unroll
      for (int r = 0; r < 8; ++r) {
        // Duplicate (y,x): unspecified winner, matches jnp .at[].set semantics.
        bev[cellRow[r] + col] = c[r] + bias2[nt];
      }
    }
  }
}

extern "C" void kernel_launch(void* const* d_in, const int* in_sizes, int n_in,
                              void* d_out, int out_size, void* d_ws, size_t ws_size,
                              hipStream_t stream) {
  const float* pos   = (const float*)d_in[0];
  const float* obj   = (const float*)d_in[1];
  const float* w1    = (const float*)d_in[2];
  const float* b1    = (const float*)d_in[3];
  const float* w2    = (const float*)d_in[4];
  const float* b2    = (const float*)d_in[5];
  const int*   grids = (const int*)d_in[6];
  const int*   ids   = (const int*)d_in[7];
  float*       bev   = (float*)d_out;

  // 1) zero-fill 1024*1024*64 f32 output (harness poisons it).
  const int n4 = (HGT * WID * OUT_DIM) / 4;          // 16,777,216 float4
  bev_zero_kernel<<<n4 / 256, 256, 0, stream>>>((float4*)bev, n4);

  // 2) fused gather + MLP (WMMA) + scatter; 8 tiles per wave amortize the
  //    per-block weight staging.
  const int nblocks = NPOINTS / PTS_PER_BLOCK;       // 2048
  bev_wmma_kernel<<<nblocks, TPB, 65536, stream>>>(pos, obj, w1, b1, w2, b2,
                                                   grids, ids, bev);
}
